// SimpleModel1_44753559224986
// MI455X (gfx1250) — matrix operand order repaired and checked by reading, not compiled, not run on hardware
//
#include <hip/hip_runtime.h>
#include <hip/hip_bf16.h>

// ---------------------------------------------------------------------------
// Fused LSTM + FC for gfx1250 (MI455X).
//   x:[64,2048,128] f32, W_ih:[128,512], W_hh:[128,512], b:[512],
//   fc_w:[2,262144], fc_b:[2]  ->  out:[64,2] f32
//
// 4 workgroups x 256 threads (8 waves). WG g owns batch rows [16g,16g+16).
// All weights resident in LDS as f16; gates computed with
// v_wmma_f32_16x16x32_f16; recurrence state c in registers; final FC
// accumulated on the fly (no hs buffer). Bias is pre-splatted into a
// persistent accumulator image and consumed as the C-operand of the first
// WMMA of each step (kc==0 peeled) so no per-step splat movs appear on the
// serial critical path.
// ---------------------------------------------------------------------------

typedef __attribute__((ext_vector_type(16))) _Float16 v16h;
typedef __attribute__((ext_vector_type(8)))  _Float16 v8h;
typedef __attribute__((ext_vector_type(8)))  float    v8f;

union V16 { v16h v; v8h h[2]; };

// problem sizes
constexpr int TT = 2048;   // sequence length (N_SEQ)
constexpr int DD = 128;    // input feature dim (SEQ_LEN)
constexpr int HH = 128;    // hidden
constexpr int GG = 512;    // 4*H

// LDS layout (halves / floats, padded strides for bank spreading)
constexpr int SW  = 264;   // W row stride (halves) per gate-column n
constexpr int SX  = 264;   // xh row stride (halves) per batch row
constexpr int SG  = 516;   // gates row stride (floats)
constexpr int SFW = 132;   // staged fc_w row stride (floats)

constexpr size_t OFF_W = 0;                              // 512*264*2 = 270336
constexpr size_t OFF_X = OFF_W + (size_t)GG * SW * 2;    // xh: 16*264*2 = 8448
constexpr size_t OFF_G = OFF_X + (size_t)16 * SX * 2;    // gates: 16*516*4 = 33024
constexpr size_t OFF_F = OFF_G + (size_t)16 * SG * 4;    // fcw: 2*132*4 = 1056
constexpr size_t SMEM_BYTES = OFF_F + (size_t)2 * SFW * 4;  // = 312864 < 320KB

__device__ __forceinline__ float sigmoidf_fast(float x) {
    return 1.0f / (1.0f + __expf(-x));
}

__global__ __launch_bounds__(256, 1)
void lstm_fused_kernel(const float* __restrict__ x,
                       const float* __restrict__ h0,
                       const float* __restrict__ c0,
                       const float* __restrict__ Wih,
                       const float* __restrict__ Whh,
                       const float* __restrict__ bias,
                       const float* __restrict__ fcw,
                       const float* __restrict__ fcb,
                       float* __restrict__ out)
{
    extern __shared__ char smem[];
    _Float16* sW = (_Float16*)(smem + OFF_W);  // sW[n*SW + k], k<128: W_ih, k>=128: W_hh
    _Float16* sX = (_Float16*)(smem + OFF_X);  // sX[r*SX + k], k<128: x_t, k>=128: h_{t-1}
    float*    sG = (float*)   (smem + OFF_G);  // activated gates [16][512]
    float*    sF = (float*)   (smem + OFF_F);  // staged fc_w rows [2][128]

    const int tid   = threadIdx.x;
    const int lane  = tid & 31;
    const int w     = tid >> 5;          // wave 0..7
    const int l16   = lane & 15;
    const int bbase = blockIdx.x * 16;   // first batch row of this WG

    // ---- load [W_ih ; W_hh] -> LDS f16, transposed to sW[n][k] ----
    for (int idx = tid; idx < DD * GG; idx += 256) {
        const int k = idx >> 9;          // 0..127 (row of W in memory)
        const int n = idx & 511;         // gate column
        sW[(size_t)n * SW + k]       = (_Float16)Wih[idx];
        sW[(size_t)n * SW + 128 + k] = (_Float16)Whh[idx];
    }

    // ---- elementwise thread mapping: (row er, col ec + 16q) ----
    const int er = tid >> 4;             // 0..15 batch row within tile
    const int ec = tid & 15;
    float creg[8];                       // cell state lives in registers
    float facc0 = 0.0f, facc1 = 0.0f;    // fc accumulators
    #pragma unroll
    for (int q = 0; q < 8; ++q) {
        const int h = ec + 16 * q;
        creg[q] = c0[(size_t)(bbase + er) * HH + h];
        sX[(size_t)er * SX + 128 + h] = (_Float16)h0[(size_t)(bbase + er) * HH + h];
    }

    // ---- persistent bias accumulator image for this wave's 4 N-tiles ----
    // C/D layout: element v (all 8) of lane L belongs to column (ntile*16+l16),
    // and bias depends only on the column -> splat once, reuse every step as
    // the C operand of the peeled first WMMA.
    v8f biasv[4];
    #pragma unroll
    for (int j = 0; j < 4; ++j) {
        const float bj = bias[(w * 4 + j) * 16 + l16];
        v8f a;
        #pragma unroll
        for (int v = 0; v < 8; ++v) a[v] = bj;
        biasv[j] = a;
    }

    // ---- x streaming: thread loads 8 contiguous floats of its row ----
    const int xr  = er;
    const int xd0 = ec * 8;
    const float* xrow = x + (size_t)(bbase + xr) * TT * DD + xd0;
    float4 xa = *(const float4*)(xrow);
    float4 xb = *(const float4*)(xrow + 4);
    {   // commit x_0 to LDS (one b128 store)
        v8h hx;
        hx[0]=(_Float16)xa.x; hx[1]=(_Float16)xa.y; hx[2]=(_Float16)xa.z; hx[3]=(_Float16)xa.w;
        hx[4]=(_Float16)xb.x; hx[5]=(_Float16)xb.y; hx[6]=(_Float16)xb.z; hx[7]=(_Float16)xb.w;
        *(v8h*)(sX + (size_t)xr * SX + xd0) = hx;
    }
    __syncthreads();

    // ---- WMMA fragment base pointers (contiguous k = 16*(lane/16)+s, same order for A and B) ----
    // A (16x32, MxK): lane<16 holds row=lane, k = k0+0..15, k0=kc*32
    //                 lane>=16 holds row=lane-16, k = k0+16..31
    const int arow  = l16;
    const int kAoff = (lane < 16) ? 0 : 16;
    const _Float16* pA = sX + (size_t)arow * SX + kAoff;
    // B (32x16, KxN): lane<16 holds col n, k = kc*32 + 0..15 (contiguous)
    //                 lane>=16 holds col n, k = kc*32 + 16..31
    const int kBoff = (lane < 16) ? 0 : 16;
    const _Float16* pB[4];
    #pragma unroll
    for (int j = 0; j < 4; ++j) {
        const int nt = w * 4 + j;
        pB[j] = sW + (size_t)(nt * 16 + l16) * SW + kBoff;
    }
    const bool isTanhWave = (w == 4) || (w == 5);   // cols [256,384) = g-gate

    for (int t = 0; t < TT; ++t) {
        // stage fc_w rows for this t (2x128 distinct values, one load/thread)
        {
            const int j = tid >> 7;          // 0..1
            const int h = tid & 127;
            sF[j * SFW + h] = fcw[(size_t)j * (TT * HH) + (size_t)t * HH + h];
        }
        // prefetch x_{t+1} into registers; hint x_{t+2} toward L2
        float4 nxa = xa, nxb = xb;
        if (t + 1 < TT) {
            nxa = *(const float4*)(xrow + (size_t)(t + 1) * DD);
            nxb = *(const float4*)(xrow + (size_t)(t + 1) * DD + 4);
        }
        if (t + 2 < TT) __builtin_prefetch(xrow + (size_t)(t + 2) * DD, 0, 1);

        // ---- gates = [x_t ; h] @ [W_ih ; W_hh] + b  via WMMA f16->f32 ----
        v8f acc[4];
        {   // kc == 0 peeled: C operand = persistent bias image (no splat movs)
            const _Float16* a0 = pA;
            V16 af;
            af.h[0] = *(const v8h*)(a0);
            af.h[1] = *(const v8h*)(a0 + 8);
            #pragma unroll
            for (int j = 0; j < 4; ++j) {
                const _Float16* b0 = pB[j];
                V16 bf;
                bf.h[0] = *(const v8h*)(b0);
                bf.h[1] = *(const v8h*)(b0 + 8);
                acc[j] = __builtin_amdgcn_wmma_f32_16x16x32_f16(
                    false, af.v, false, bf.v, (short)0, biasv[j], false, false);
            }
        }
        #pragma unroll
        for (int kc = 1; kc < 8; ++kc) {
            const _Float16* a0 = pA + kc * 32;
            V16 af;
            af.h[0] = *(const v8h*)(a0);
            af.h[1] = *(const v8h*)(a0 + 8);
            #pragma unroll
            for (int j = 0; j < 4; ++j) {
                const _Float16* b0 = pB[j] + kc * 32;
                V16 bf;
                bf.h[0] = *(const v8h*)(b0);
                bf.h[1] = *(const v8h*)(b0 + 8);
                acc[j] = __builtin_amdgcn_wmma_f32_16x16x32_f16(
                    false, af.v, false, bf.v, (short)0, acc[j], false, false);
            }
        }

        // ---- activate (wave-uniform gate type) and publish to LDS ----
        const int rb = (lane >= 16) ? 8 : 0;
        #pragma unroll
        for (int j = 0; j < 4; ++j) {
            const int col = (w * 4 + j) * 16 + l16;
            #pragma unroll
            for (int v = 0; v < 8; ++v) {
                const float gv = acc[j][v];
                const float av = isTanhWave ? tanhf(gv) : sigmoidf_fast(gv);
                sG[(size_t)(rb + v) * SG + col] = av;
            }
        }
        __syncthreads();

        // ---- elementwise cell update + fused FC accumulation ----
        #pragma unroll
        for (int q = 0; q < 8; ++q) {
            const int h = ec + 16 * q;
            const float iv = sG[(size_t)er * SG + h];
            const float fv = sG[(size_t)er * SG + 128 + h];
            const float gv = sG[(size_t)er * SG + 256 + h];
            const float ov = sG[(size_t)er * SG + 384 + h];
            const float cn = fv * creg[q] + iv * gv;
            creg[q] = cn;
            const float hn = ov * tanhf(cn);
            sX[(size_t)er * SX + 128 + h] = (_Float16)hn;   // h_t for next step
            facc0 += hn * sF[h];
            facc1 += hn * sF[SFW + h];
        }
        // commit x_{t+1} (x-region free: all WMMA reads done before barrier)
        {
            v8h hx;
            hx[0]=(_Float16)nxa.x; hx[1]=(_Float16)nxa.y; hx[2]=(_Float16)nxa.z; hx[3]=(_Float16)nxa.w;
            hx[4]=(_Float16)nxb.x; hx[5]=(_Float16)nxb.y; hx[6]=(_Float16)nxb.z; hx[7]=(_Float16)nxb.w;
            *(v8h*)(sX + (size_t)xr * SX + xd0) = hx;
            xa = nxa; xb = nxb;
        }
        __syncthreads();
    }

    // ---- final FC reduction: 16 partials per batch row (reuse gate LDS) ----
    float* red = sG;
    red[tid * 2 + 0] = facc0;
    red[tid * 2 + 1] = facc1;
    __syncthreads();
    if (tid < 16) {
        float s0 = 0.0f, s1 = 0.0f;
        #pragma unroll
        for (int i = 0; i < 16; ++i) {
            s0 += red[(tid * 16 + i) * 2 + 0];
            s1 += red[(tid * 16 + i) * 2 + 1];
        }
        out[(size_t)(bbase + tid) * 2 + 0] = s0 + fcb[0];
        out[(size_t)(bbase + tid) * 2 + 1] = s1 + fcb[1];
    }
}

extern "C" void kernel_launch(void* const* d_in, const int* in_sizes, int n_in,
                              void* d_out, int out_size, void* d_ws, size_t ws_size,
                              hipStream_t stream) {
    const float* x   = (const float*)d_in[0];
    const float* h0  = (const float*)d_in[1];
    const float* c0  = (const float*)d_in[2];
    const float* Wih = (const float*)d_in[3];
    const float* Whh = (const float*)d_in[4];
    const float* b   = (const float*)d_in[5];
    const float* fcw = (const float*)d_in[6];
    const float* fcb = (const float*)d_in[7];
    float* out = (float*)d_out;

    // allow >64KB dynamic LDS (gfx1250: up to 320KB per workgroup)
    (void)hipFuncSetAttribute((const void*)lstm_fused_kernel,
                              hipFuncAttributeMaxDynamicSharedMemorySize,
                              (int)SMEM_BYTES);

    lstm_fused_kernel<<<dim3(4), dim3(256), SMEM_BYTES, stream>>>(
        x, h0, c0, Wih, Whh, b, fcw, fcb, out);
}
